// TransformerLayer_69037304315981
// MI455X (gfx1250) — compile-verified, matrix operand order checked by reading, not run on hardware
//
#include <hip/hip_runtime.h>

typedef unsigned short ushort_t;
typedef unsigned int   uint_t;
typedef __attribute__((ext_vector_type(16))) __bf16 v16bf;
typedef __attribute__((ext_vector_type(8)))  float  v8f;

#define BM 128
#define BN 128
#define BK 32
#define LSTR 40   // padded LDS row stride in ushorts: 80B -> 16B aligned rows, 20-bank stride

__device__ __forceinline__ ushort_t f2bf(float f) {
  uint_t u = __float_as_uint(f);
  u += 0x7FFFu + ((u >> 16) & 1u);     // round-to-nearest-even
  return (ushort_t)(u >> 16);
}

// CDNA5 async global->LDS copy, 16 bytes per lane (GLOBAL_LOAD_ASYNC_TO_LDS_B128,
// tracked by ASYNCcnt). vdst = LDS byte address (flat addr low 32 bits per ISA
// aperture mapping), vaddr = 64-bit global address.
__device__ __forceinline__ void async_copy_b128(const void* lds_ptr, const void* gptr) {
  uint_t laddr = (uint_t)(unsigned long long)lds_ptr;
  asm volatile("global_load_async_to_lds_b128 %0, %1, off"
               :: "v"(laddr), "v"(gptr)
               : "memory");
}
__device__ __forceinline__ void wait_asynccnt0() {
  asm volatile("s_wait_asynccnt 0x0" ::: "memory");
}

union FragAB { v16bf v; uint_t w[8]; };

// ---------------------------------------------------------------------------
// Tiled bf16 WMMA GEMM: C[M,N] = A[M,K] @ B[K,N] (+epilogue)
// AKIND: 0 = A is bf16 row-major (async-to-LDS); 1 = A is f32 (convert on load)
// TRANSB: false = B is [K,N] row-major (scatter transpose + prefetch);
//         true  = B source is [N,K] row-major (async-to-LDS straight copy)
// EPI: 0 = scale + causal mask (col>row -> -inf), f32 out   (scores)
//      1 = +bias, ReLU, bf16 out                            (FFN1)
//      2 = +bias, bf16 out                                  (QKV)
//      3 = +bias, f32 out                                   (FFN2)
//      4 = plain f32 out                                    (atten @ v)
// Batched via blockIdx.z with element strides sA/sB/sC (0 for unbatched operands).
// ---------------------------------------------------------------------------
template <int AKIND, bool TRANSB, int EPI>
__global__ __launch_bounds__(128) void gemm128_wmma(
    const void* __restrict__ Aptr, const ushort_t* __restrict__ Bsrc,
    float* __restrict__ Cf, ushort_t* __restrict__ Cbf,
    const float* __restrict__ bias,
    int K, int lda, int ldb, int ldc,
    long long sA, long long sB, long long sC, float scale) {
  __shared__ __align__(16) ushort_t lA[BM * LSTR];
  __shared__ __align__(16) ushort_t lB[BN * LSTR];

  const int tid  = threadIdx.x;
  const int lane = tid & 31;
  const int wave = tid >> 5;
  const int wm   = (wave >> 1) * 64;
  const int wn   = (wave & 1) * 64;
  const int m0   = blockIdx.y * BM;
  const int n0   = blockIdx.x * BN;
  const long long b = blockIdx.z;

  const ushort_t* Abf = (AKIND == 0) ? ((const ushort_t*)Aptr + b * sA) : nullptr;
  const float*    Af  = (AKIND == 1) ? ((const float*)Aptr + b * sA) : nullptr;
  const ushort_t* Bp  = Bsrc + b * sB;

  const int hf  = lane >> 4;   // K-half select per documented 16-bit layouts
  const int l15 = lane & 15;

  v8f acc[4][4];
  const v8f vzero = {0.f, 0.f, 0.f, 0.f, 0.f, 0.f, 0.f, 0.f};
#pragma unroll
  for (int i = 0; i < 4; ++i)
#pragma unroll
    for (int j = 0; j < 4; ++j) acc[i][j] = vzero;

  for (int k0 = 0; k0 < K; k0 += BK) {
    __syncthreads();
    // ---- A tile -> lA[row][k], 128x32 ----
    if constexpr (AKIND == 0) {
#pragma unroll
      for (int p = 0; p < 4; ++p) {
        int idx = tid + p * 128, row = idx >> 2, seg = idx & 3;
        const ushort_t* g = Abf + (long long)(m0 + row) * lda + k0 + seg * 8;
        async_copy_b128(&lA[row * LSTR + seg * 8], g);
      }
    } else {
#pragma unroll
      for (int p = 0; p < 4; ++p) {
        int idx = tid + p * 128, row = idx >> 2, seg = idx & 3;
        const float* src = Af + (long long)(m0 + row) * lda + k0 + seg * 8;
        float4 f0 = *(const float4*)(src);
        float4 f1 = *(const float4*)(src + 4);
        union { ushort_t u[8]; uint4 q; } pk;
        pk.u[0] = f2bf(f0.x); pk.u[1] = f2bf(f0.y); pk.u[2] = f2bf(f0.z); pk.u[3] = f2bf(f0.w);
        pk.u[4] = f2bf(f1.x); pk.u[5] = f2bf(f1.y); pk.u[6] = f2bf(f1.z); pk.u[7] = f2bf(f1.w);
        *(uint4*)&lA[row * LSTR + seg * 8] = pk.q;
      }
    }
    // ---- B tile -> lB[n][k], 128x32 ----
    if constexpr (TRANSB) {
      // B[k][n] = src[n][k]: rows of src are contiguous in k -> async straight copy
#pragma unroll
      for (int p = 0; p < 4; ++p) {
        int idx = tid + p * 128, n = idx >> 2, seg = idx & 3;
        const ushort_t* g = Bp + (long long)(n0 + n) * ldb + k0 + seg * 8;
        async_copy_b128(&lB[n * LSTR + seg * 8], g);
      }
    } else {
      // src is [K,N]: coalesced read along n, scatter-transpose into lB[n][k]
#pragma unroll
      for (int p = 0; p < 4; ++p) {
        int idx = tid + p * 128, kk = idx >> 4, seg = idx & 15;
        uint4 d = *(const uint4*)(Bp + (long long)(k0 + kk) * ldb + n0 + seg * 8);
        if (k0 + BK < K)   // prefetch next K-tile (global_prefetch_b8)
          __builtin_prefetch(Bp + (long long)(k0 + BK + kk) * ldb + n0 + seg * 8, 0, 1);
        const ushort_t* e = (const ushort_t*)&d;
#pragma unroll
        for (int q = 0; q < 8; ++q) lB[(seg * 8 + q) * LSTR + kk] = e[q];
      }
    }
    if constexpr (AKIND == 0 || TRANSB) wait_asynccnt0();
    __syncthreads();

    // ---- fragments + 16 WMMAs ----
    FragAB afr[4], bfr[4];
#pragma unroll
    for (int mt = 0; mt < 4; ++mt) {
      const int row = wm + mt * 16 + l15;
#pragma unroll
      for (int j = 0; j < 8; ++j) {
        int kb = ((j < 4) ? 0 : 16) + hf * 8 + (j & 3) * 2;
        afr[mt].w[j] = *(const uint_t*)&lA[row * LSTR + kb];
      }
    }
#pragma unroll
    for (int nt = 0; nt < 4; ++nt) {
      const int col = wn + nt * 16 + l15;
#pragma unroll
      for (int j = 0; j < 8; ++j) {
        int kb = ((j < 4) ? 0 : 16) + hf * 8 + (j & 3) * 2;
        bfr[nt].w[j] = *(const uint_t*)&lB[col * LSTR + kb];
      }
    }
#pragma unroll
    for (int mt = 0; mt < 4; ++mt)
#pragma unroll
      for (int nt = 0; nt < 4; ++nt)
        acc[mt][nt] = __builtin_amdgcn_wmma_f32_16x16x32_bf16(
            false, afr[mt].v, false, bfr[nt].v, (short)0, acc[mt][nt], false, false);
  }

  // ---- epilogue: C/D layout: lanes 0-15 -> M=e, lanes 16-31 -> M=e+8; N = lane&15 ----
  (void)scale;
#pragma unroll
  for (int mt = 0; mt < 4; ++mt) {
#pragma unroll
    for (int nt = 0; nt < 4; ++nt) {
      const int ng = n0 + wn + nt * 16 + l15;
      const int mb = m0 + wm + mt * 16 + 8 * hf;
      float bv = 0.f;
      if constexpr (EPI == 1 || EPI == 2 || EPI == 3) bv = bias[ng];
#pragma unroll
      for (int e = 0; e < 8; ++e) {
        const int mg = mb + e;
        float val = acc[mt][nt][e];
        const long long co = b * sC + (long long)mg * ldc + ng;
        if constexpr (EPI == 0) {
          val *= scale;
          if (ng > mg) val = -__builtin_inff();
          Cf[co] = val;
        } else if constexpr (EPI == 1) {
          Cbf[co] = f2bf(fmaxf(val + bv, 0.f));
        } else if constexpr (EPI == 2) {
          Cbf[co] = f2bf(val + bv);
        } else if constexpr (EPI == 3) {
          Cf[co] = val + bv;
        } else {
          Cf[co] = val;
        }
      }
    }
  }
}

// ---------------------------------------------------------------------------
// Softmax over dim=1 (query axis): one thread per (b, key-column), online max/sum.
// ---------------------------------------------------------------------------
__global__ __launch_bounds__(256) void softmax_dim1_kernel(float* __restrict__ atten) {
  const int S = 2048;
  const long long b = blockIdx.y;
  const int c = blockIdx.x * 256 + threadIdx.x;
  float* base = atten + b * (long long)S * S + c;
  const float ninf = -__builtin_inff();
  float m = ninf, s = 0.f;
  for (int q = 0; q < S; ++q) {
    float x = base[(long long)q * S];
    if (x == ninf) continue;
    if (x > m) { s = s * __expf(m - x) + 1.f; m = x; }
    else       { s += __expf(x - m); }
  }
  const float inv = 1.f / s;
  for (int q = 0; q < S; ++q) {
    float x = base[(long long)q * S];
    base[(long long)q * S] = (x == ninf) ? 0.f : __expf(x - m) * inv;
  }
}

// ---------------------------------------------------------------------------
// Fused residual-add + LayerNorm over H=1024. Writes f32 and (optionally) bf16.
// ---------------------------------------------------------------------------
__global__ __launch_bounds__(256) void add_ln_kernel(
    const float* __restrict__ X, const float* __restrict__ R,
    const float* __restrict__ g, const float* __restrict__ be,
    float* __restrict__ Of, ushort_t* __restrict__ Obf) {
  const int Hc = 1024;
  const long long row = blockIdx.x;
  const float* x = X + row * Hc;
  const float* r = R + row * Hc;
  const int t = threadIdx.x;
  float a[4];
  float s = 0.f, ss = 0.f;
#pragma unroll
  for (int i = 0; i < 4; ++i) {
    int idx = t + i * 256;
    float v = x[idx] + r[idx];
    a[i] = v; s += v; ss += v * v;
  }
  __shared__ float red[256];
  red[t] = s; __syncthreads();
  for (int o = 128; o > 0; o >>= 1) { if (t < o) red[t] += red[t + o]; __syncthreads(); }
  const float mean = red[0] * (1.f / 1024.f);
  __syncthreads();
  red[t] = ss; __syncthreads();
  for (int o = 128; o > 0; o >>= 1) { if (t < o) red[t] += red[t + o]; __syncthreads(); }
  const float var = red[0] * (1.f / 1024.f) - mean * mean;
  const float rstd = rsqrtf(var + 1e-5f);
#pragma unroll
  for (int i = 0; i < 4; ++i) {
    int idx = t + i * 256;
    float o = (a[i] - mean) * rstd * g[idx] + be[idx];
    Of[row * Hc + idx] = o;
    if (Obf) Obf[row * Hc + idx] = f2bf(o);
  }
}

__global__ void cvt_bf16_kernel(const float* __restrict__ in, ushort_t* __restrict__ out,
                                long long n) {
  long long i = (long long)blockIdx.x * blockDim.x + threadIdx.x;
  if (i < n) out[i] = f2bf(in[i]);
}

__global__ void lengths_kernel(const int* __restrict__ len, float* __restrict__ out) {
  int i = threadIdx.x;
  if (i < 8) out[i] = (float)len[i];
}

// ---------------------------------------------------------------------------
extern "C" void kernel_launch(void* const* d_in, const int* in_sizes, int n_in,
                              void* d_out, int out_size, void* d_ws, size_t ws_size,
                              hipStream_t stream) {
  (void)in_sizes; (void)n_in; (void)out_size; (void)ws_size;
  const int B = 8, S = 2048, H = 1024;
  const long long BSH = (long long)B * S * H;   // 16,777,216
  const long long HH  = (long long)H * H;

  const float* x   = (const float*)d_in[0];
  const int*   len = (const int*)d_in[1];
  const float* Wq = (const float*)d_in[3];  const float* bq = (const float*)d_in[4];
  const float* Wk = (const float*)d_in[5];  const float* bk = (const float*)d_in[6];
  const float* Wv = (const float*)d_in[7];  const float* bv = (const float*)d_in[8];
  const float* W1 = (const float*)d_in[9];  const float* b1 = (const float*)d_in[10];
  const float* W2 = (const float*)d_in[11]; const float* b2 = (const float*)d_in[12];
  const float* g1 = (const float*)d_in[13]; const float* be1 = (const float*)d_in[14];
  const float* g2 = (const float*)d_in[15]; const float* be2 = (const float*)d_in[16];

  // outputs: (y[B,S,H], lengths[B], atten_w[B,S,S]) flattened
  float* y      = (float*)d_out;
  float* lenOut = y + BSH;
  float* atten  = lenOut + B;   // softmax computed in place here

  // workspace layout (~280 MB)
  ushort_t* xb  = (ushort_t*)d_ws;
  ushort_t* qb  = xb + BSH;
  ushort_t* kbm = qb + BSH;
  ushort_t* vb  = kbm + BSH;
  ushort_t* wqb = vb + BSH;
  ushort_t* wkb = wqb + HH;
  ushort_t* wvb = wkb + HH;
  ushort_t* w1b = wvb + HH;
  ushort_t* w2b = w1b + HH;
  float* attnout = (float*)(w2b + HH);   // reused as FFN2 output after LN1
  float* o1      = attnout + BSH;
  ushort_t* o1b  = qb;    // q no longer needed after scores
  ushort_t* h1b  = kbm;   // k no longer needed after scores
  float* midf    = attnout;

  // --- converts ---
  {
    int blk = 256;
    cvt_bf16_kernel<<<(unsigned)((BSH + blk - 1) / blk), blk, 0, stream>>>(x, xb, BSH);
    cvt_bf16_kernel<<<(unsigned)((HH + blk - 1) / blk), blk, 0, stream>>>(Wq, wqb, HH);
    cvt_bf16_kernel<<<(unsigned)((HH + blk - 1) / blk), blk, 0, stream>>>(Wk, wkb, HH);
    cvt_bf16_kernel<<<(unsigned)((HH + blk - 1) / blk), blk, 0, stream>>>(Wv, wvb, HH);
    cvt_bf16_kernel<<<(unsigned)((HH + blk - 1) / blk), blk, 0, stream>>>(W1, w1b, HH);
    cvt_bf16_kernel<<<(unsigned)((HH + blk - 1) / blk), blk, 0, stream>>>(W2, w2b, HH);
  }
  lengths_kernel<<<1, 32, 0, stream>>>(len, lenOut);

  const dim3 tb(128);
  // --- QKV projections: [16384,1024] = x @ W + b, bf16 out ---
  {
    dim3 g(H / BN, (B * S) / BM, 1);
    gemm128_wmma<0, false, 2><<<g, tb, 0, stream>>>(xb, wqb, nullptr, qb, bq,
                                                    H, H, H, H, 0, 0, 0, 0.f);
    gemm128_wmma<0, false, 2><<<g, tb, 0, stream>>>(xb, wkb, nullptr, kbm, bk,
                                                    H, H, H, H, 0, 0, 0, 0.f);
    gemm128_wmma<0, false, 2><<<g, tb, 0, stream>>>(xb, wvb, nullptr, vb, bv,
                                                    H, H, H, H, 0, 0, 0, 0.f);
  }
  // --- scores = q @ kᵀ * 1/√H, causal mask, f32 -> atten (in d_out) ---
  {
    dim3 g(S / BN, S / BM, B);
    gemm128_wmma<0, true, 0><<<g, tb, 0, stream>>>(qb, kbm, atten, nullptr, nullptr,
                                                   H, H, H, S,
                                                   (long long)S * H, (long long)S * H,
                                                   (long long)S * S, 1.f / 32.f);
  }
  // --- softmax over query axis (dim=1), in place ---
  {
    dim3 g(S / 256, B);
    softmax_dim1_kernel<<<g, 256, 0, stream>>>(atten);
  }
  // --- out = atten @ v (A is f32, converted on load), f32 -> attnout ---
  {
    dim3 g(H / BN, S / BM, B);
    gemm128_wmma<1, false, 4><<<g, tb, 0, stream>>>(atten, vb, attnout, nullptr, nullptr,
                                                    S, S, H, H,
                                                    (long long)S * S, (long long)S * H,
                                                    (long long)S * H, 0.f);
  }
  // --- o1 = LN(x + attnout) ---
  add_ln_kernel<<<B * S, 256, 0, stream>>>(x, attnout, g1, be1, o1, o1b);
  // --- FFN ---
  {
    dim3 g(H / BN, (B * S) / BM, 1);
    gemm128_wmma<0, false, 1><<<g, tb, 0, stream>>>(o1b, w1b, nullptr, h1b, b1,
                                                    H, H, H, H, 0, 0, 0, 0.f);
    gemm128_wmma<0, false, 3><<<g, tb, 0, stream>>>(h1b, w2b, midf, nullptr, b2,
                                                    H, H, H, H, 0, 0, 0, 0.f);
  }
  // --- y = LN(mid + o1) ---
  add_ln_kernel<<<B * S, 256, 0, stream>>>(midf, o1, g2, be2, y, nullptr);
}
